// GroupQueryAttention_38474317038263
// MI455X (gfx1250) — matrix operand order repaired and checked by reading, not compiled, not run on hardware
//
#include <hip/hip_runtime.h>

typedef unsigned short ushort_t;
typedef __attribute__((ext_vector_type(16))) __bf16 bf16x16;
typedef __attribute__((ext_vector_type(8)))  float  floatx8;

union Frag { unsigned int u[8]; bf16x16 v; };

// ---------- helpers ----------

// fp32 -> bf16 round-to-nearest-even (ignoring NaN edge cases)
__device__ inline ushort_t f2bf(float x) {
    union { float f; unsigned int u; } c; c.f = x;
    unsigned int u = c.u + 0x7fffu + ((c.u >> 16) & 1u);
    return (ushort_t)(u >> 16);
}

// K index of element-pair p for the 16-bit A fragment (16x32), per ISA table:
// V0: lanes0-15 K=0,1 / lanes16-31 K=8,9 ; V4: K=16,17 / 24,25 ...
__device__ inline int a_k(int lane, int p) {
    int base = (lane & 16) ? 8 : 0;
    return base + ((p < 4) ? (2 * p) : (16 + 2 * (p - 4)));
}
// K index of element-pair p for the 16-bit B fragment (32x16):
// lane = column; same K order as A: lanes0-15 K=0..7,16..23 / lanes16-31 K=8..15,24..31
__device__ inline int b_k(int lane, int p) {
    return ((lane & 16) ? 8 : 0) + ((p < 4) ? (2 * p) : (16 + 2 * (p - 4)));
}

// Build a WMMA fragment from an LDS tile where the fragment's K dimension is
// CONTIGUOUS in the tile row (compiler merges these into ds_load_b128).
template <bool BMAP>
__device__ inline bf16x16 load_frag(const ushort_t* buf, int ld, int row, int kbase) {
    int lane = threadIdx.x & 31;
    const ushort_t* base = buf + (row + (lane & 15)) * ld + kbase;
    Frag f;
#pragma unroll
    for (int p = 0; p < 8; ++p) {
        int k = BMAP ? b_k(lane, p) : a_k(lane, p);
        f.u[p] = *(const unsigned int*)(base + k);
    }
    return f.v;
}

__device__ inline floatx8 wmma_bf16(bf16x16 a, bf16x16 b, floatx8 c) {
    return __builtin_amdgcn_wmma_f32_16x16x32_bf16(false, a, false, b, (short)0, c,
                                                   false, false);
}

// Async DMA: 16 bytes per lane, global -> LDS, tracked by ASYNCcnt.
// GVS mode: SGPR 64-bit base + per-lane 32-bit byte offset; VDST = per-lane
// LDS byte offset (low 32 bits of the generic pointer == LDS offset per the
// aperture mapping).
__device__ inline void async_copy16(const void* gbase, unsigned voff, void* ldsdst) {
    unsigned lofs = (unsigned)(uintptr_t)ldsdst;
    asm volatile("global_load_async_to_lds_b128 %0, %1, %2"
                 :: "v"(lofs), "v"(voff), "s"(gbase)
                 : "memory");
}
__device__ inline void wait_async0() {
#if __has_builtin(__builtin_amdgcn_s_wait_asynccnt)
    __builtin_amdgcn_s_wait_asynccnt(0);
#else
    asm volatile("s_wait_asynccnt 0x0" ::: "memory");
#endif
}

// ---------- kernels ----------

__global__ __launch_bounds__(256) void cvt_f32_bf16(const float* __restrict__ in,
                                                    ushort_t* __restrict__ out, int n) {
    int i = blockIdx.x * blockDim.x + threadIdx.x;
    int stride = gridDim.x * blockDim.x;
    for (; i < n; i += stride) out[i] = f2bf(in[i]);
}

// C[M,N] = A[M,K](bf16) * W[K,N](bf16) + bias ; out bf16 or fp32.
// Block = 256 threads (8 waves, 4x2), tile 128x128, K-step 32.
// Each wave: 32 rows x 64 cols = 8 WMMAs per K-step from 6 fragment loads.
template <bool OUT_BF16>
__global__ __launch_bounds__(256, 1)
void gemm_tn(const ushort_t* __restrict__ A, const ushort_t* __restrict__ W,
             const float* __restrict__ bias, void* __restrict__ Cout,
             int M, int N, int K) {
    constexpr int LDT = 40;                  // 32 + 8 pad (ushorts)
    __shared__ ushort_t As[128 * LDT];       // [m][k] row-major
    __shared__ ushort_t Bs[128 * LDT];       // W tile stored TRANSPOSED: [n][k]
    const int tid = threadIdx.x;
    const int lane = tid & 31;
    const int wave = tid >> 5;               // 0..7
    const int wr = wave >> 1;                // 0..3 -> 32-row slab
    const int wc = wave & 1;                 // 0..1 -> 64-col slab
    const int m0 = blockIdx.y * 128;
    const int n0 = blockIdx.x * 128;

    floatx8 acc[2][4] = {};

    for (int k0 = 0; k0 < K; k0 += 32) {
        // async DMA stage A: 128 rows x 32 cols in 16B chunks
        for (int i = tid; i < 512; i += 256) {
            int r = i >> 2, c = (i & 3) * 8;
            unsigned voff = (unsigned)((((size_t)(m0 + r)) * K + k0 + c) * 2);
            async_copy16(A, voff, &As[r * LDT + c]);
        }
        // stage W transposed: W[k0+k][n0+c..c+7] -> Bs[c+j][k]
        for (int i = tid; i < 512; i += 256) {
            int k = i >> 4, c = (i & 15) * 8;
            uint4 d = *(const uint4*)&W[(size_t)(k0 + k) * N + n0 + c];
            const ushort_t* s = (const ushort_t*)&d;
#pragma unroll
            for (int j = 0; j < 8; ++j) Bs[(c + j) * LDT + k] = s[j];
        }
        wait_async0();
        __syncthreads();

        bf16x16 af0 = load_frag<false>(As, LDT, wr * 32 + 0,  0);
        bf16x16 af1 = load_frag<false>(As, LDT, wr * 32 + 16, 0);
#pragma unroll
        for (int nt = 0; nt < 4; ++nt) {
            bf16x16 bfr = load_frag<true>(Bs, LDT, wc * 64 + nt * 16, 0);
            acc[0][nt] = wmma_bf16(af0, bfr, acc[0][nt]);
            acc[1][nt] = wmma_bf16(af1, bfr, acc[1][nt]);
        }
        __syncthreads();
    }

    // epilogue: C layout lane mapping M = r + 8*(lane>>4), N = lane&15
#pragma unroll
    for (int mt = 0; mt < 2; ++mt) {
#pragma unroll
        for (int nt = 0; nt < 4; ++nt) {
            int col = n0 + wc * 64 + nt * 16 + (lane & 15);
            float bval = bias[col];
#pragma unroll
            for (int r = 0; r < 8; ++r) {
                int row = m0 + wr * 32 + mt * 16 + r + ((lane >> 4) << 3);
                float v = acc[mt][nt][r] + bval;
                if (OUT_BF16)
                    ((ushort_t*)Cout)[(size_t)row * N + col] = f2bf(v);
                else
                    ((float*)Cout)[(size_t)row * N + col] = v;
            }
        }
    }
}

// Flash attention for GQA. grid = (S/64, NUM_HEAD, B), block = 128 (4 waves).
// Each wave owns 16 query rows. Q:[B*S,2048] bf16, K/V:[B*S,512] bf16,
// O:[B*S,2048] bf16. head h uses kv group g = h/4.
__global__ __launch_bounds__(128, 1)
void gqa_attn(const ushort_t* __restrict__ Q, const ushort_t* __restrict__ Kc,
              const ushort_t* __restrict__ Vc, ushort_t* __restrict__ O) {
    constexpr int LDK = 72;                     // 64 + 8 pad
    __shared__ ushort_t Kt[64 * LDK];           // [sk][d]
    __shared__ ushort_t Vt[64 * LDK];           // transposed: [d][sk]
    __shared__ ushort_t Pt[4][16 * LDK];        // per-wave P tile [m][sk]

    const int tid = threadIdx.x, lane = tid & 31, wave = tid >> 5;
    const int qb = blockIdx.x, h = blockIdx.y, b = blockIdx.z;
    const int g = h >> 2;  // REP = 4

    // ---- Q fragments straight from global (pairs contiguous along head dim)
    bf16x16 qf[2];
    {
        const ushort_t* qr = Q + ((size_t)b * 2048 + qb * 64 + wave * 16 + (lane & 15)) * 2048
                               + h * 64;
        Frag f0, f1;
#pragma unroll
        for (int p = 0; p < 8; ++p) {
            f0.u[p] = *(const unsigned int*)(qr + a_k(lane, p));
            f1.u[p] = *(const unsigned int*)(qr + 32 + a_k(lane, p));
        }
        qf[0] = f0.v; qf[1] = f1.v;
    }

    floatx8 oacc[4] = {};
    float mrow[8], lrow[8];
#pragma unroll
    for (int r = 0; r < 8; ++r) { mrow[r] = -__builtin_inff(); lrow[r] = 0.f; }

    const float scale = 0.125f;  // 1/sqrt(64)

    for (int kb = 0; kb < 2048; kb += 64) {
        // ---- stage K tile [sk][d] (async DMA) and V tile transposed [d][sk]
        const size_t krow0 = (size_t)b * 2048 + kb;
        const ushort_t* vg = Vc + krow0 * 512 + g * 64;
        for (int i = tid; i < 512; i += 128) {
            int r = i >> 3, c = (i & 7) * 8;
            unsigned voff = (unsigned)(((krow0 + r) * 512 + g * 64 + c) * 2);
            async_copy16(Kc, voff, &Kt[r * LDK + c]);
            uint4 d = *(const uint4*)&vg[(size_t)r * 512 + c];
            const ushort_t* s = (const ushort_t*)&d;
#pragma unroll
            for (int j = 0; j < 8; ++j) Vt[(c + j) * LDK + r] = s[j];
        }
        wait_async0();
        __syncthreads();

        // ---- scores S = Q @ K^T  (B-frag: lane = key col sk, K-dim = d contiguous)
        floatx8 sacc[4];
#pragma unroll
        for (int nt = 0; nt < 4; ++nt) {
            floatx8 z = {};
            z = wmma_bf16(qf[0], load_frag<true>(Kt, LDK, nt * 16, 0),  z);
            z = wmma_bf16(qf[1], load_frag<true>(Kt, LDK, nt * 16, 32), z);
            sacc[nt] = z;
        }
#pragma unroll
        for (int nt = 0; nt < 4; ++nt)
#pragma unroll
            for (int r = 0; r < 8; ++r) sacc[nt][r] *= scale;

        // ---- online softmax (rows live in 16-lane halves; reduce with shfl_xor)
        float mnew[8], osc[8];
#pragma unroll
        for (int r = 0; r < 8; ++r) {
            float t = fmaxf(fmaxf(sacc[0][r], sacc[1][r]),
                            fmaxf(sacc[2][r], sacc[3][r]));
#pragma unroll
            for (int off = 8; off >= 1; off >>= 1)
                t = fmaxf(t, __shfl_xor(t, off, 32));
            mnew[r] = fmaxf(mrow[r], t);
            osc[r] = __expf(mrow[r] - mnew[r]);
        }
#pragma unroll
        for (int r = 0; r < 8; ++r) {
            float rs = 0.f;
#pragma unroll
            for (int nt = 0; nt < 4; ++nt) {
                float p = __expf(sacc[nt][r] - mnew[r]);
                sacc[nt][r] = p;
                rs += p;
            }
#pragma unroll
            for (int off = 8; off >= 1; off >>= 1) rs += __shfl_xor(rs, off, 32);
            lrow[r] = lrow[r] * osc[r] + rs;
            mrow[r] = mnew[r];
#pragma unroll
            for (int dt = 0; dt < 4; ++dt) oacc[dt][r] *= osc[r];
        }

        // ---- P (C-layout) -> per-wave LDS -> A-layout fragments
        ushort_t* pw = &Pt[wave][0];
#pragma unroll
        for (int nt = 0; nt < 4; ++nt)
#pragma unroll
            for (int r = 0; r < 8; ++r)
                pw[(r + ((lane >> 4) << 3)) * LDK + nt * 16 + (lane & 15)] =
                    f2bf(sacc[nt][r]);

        // ---- O += P @ V  (V B-frag: lane = d col, K-dim = sk contiguous in Vt)
#pragma unroll
        for (int dt = 0; dt < 4; ++dt) {
            oacc[dt] = wmma_bf16(load_frag<false>(pw, LDK, 0, 0),
                                 load_frag<true >(Vt, LDK, dt * 16, 0),  oacc[dt]);
            oacc[dt] = wmma_bf16(load_frag<false>(pw, LDK, 0, 32),
                                 load_frag<true >(Vt, LDK, dt * 16, 32), oacc[dt]);
        }
        __syncthreads();
    }

    // ---- normalize and store
    const size_t orow0 = (size_t)b * 2048 + qb * 64 + wave * 16;
#pragma unroll
    for (int r = 0; r < 8; ++r) {
        float inv = 1.f / lrow[r];
        int row = r + ((lane >> 4) << 3);
        size_t base = (orow0 + row) * 2048 + h * 64;
#pragma unroll
        for (int dt = 0; dt < 4; ++dt)
            O[base + dt * 16 + (lane & 15)] = f2bf(oacc[dt][r] * inv);
    }
}

// ---------- launch ----------

extern "C" void kernel_launch(void* const* d_in, const int* in_sizes, int n_in,
                              void* d_out, int out_size, void* d_ws, size_t ws_size,
                              hipStream_t stream) {
    (void)in_sizes; (void)n_in; (void)out_size; (void)ws_size;
    const float* X  = (const float*)d_in[0];
    const float* Wq = (const float*)d_in[1];
    const float* bq = (const float*)d_in[2];
    const float* Wk = (const float*)d_in[3];
    const float* bk = (const float*)d_in[4];
    const float* Wv = (const float*)d_in[5];
    const float* bv = (const float*)d_in[6];
    const float* Wo = (const float*)d_in[7];
    const float* bo = (const float*)d_in[8];
    float* out = (float*)d_out;

    const size_t M = 4096, H = 2048, KV = 512;  // M = B*S

    ushort_t* w = (ushort_t*)d_ws;
    ushort_t* Xb  = w; w += M * H;
    ushort_t* Wqb = w; w += H * H;
    ushort_t* Wkb = w; w += H * KV;
    ushort_t* Wvb = w; w += H * KV;
    ushort_t* Wob = w; w += H * H;
    ushort_t* Qb  = w; w += M * H;
    ushort_t* Kb  = w; w += M * KV;
    ushort_t* Vb  = w; w += M * KV;
    ushort_t* Ob  = w; w += M * H;

    cvt_f32_bf16<<<2048, 256, 0, stream>>>(X,  Xb,  (int)(M * H));
    cvt_f32_bf16<<<2048, 256, 0, stream>>>(Wq, Wqb, (int)(H * H));
    cvt_f32_bf16<<<512,  256, 0, stream>>>(Wk, Wkb, (int)(H * KV));
    cvt_f32_bf16<<<512,  256, 0, stream>>>(Wv, Wvb, (int)(H * KV));
    cvt_f32_bf16<<<2048, 256, 0, stream>>>(Wo, Wob, (int)(H * H));

    gemm_tn<true ><<<dim3(H / 128,  M / 128), 256, 0, stream>>>(Xb, Wqb, bq, Qb,
                                                                (int)M, (int)H, (int)H);
    gemm_tn<true ><<<dim3(KV / 128, M / 128), 256, 0, stream>>>(Xb, Wkb, bk, Kb,
                                                                (int)M, (int)KV, (int)H);
    gemm_tn<true ><<<dim3(KV / 128, M / 128), 256, 0, stream>>>(Xb, Wvb, bv, Vb,
                                                                (int)M, (int)KV, (int)H);

    gqa_attn<<<dim3(32, 32, 2), 128, 0, stream>>>(Qb, Kb, Vb, Ob);

    gemm_tn<false><<<dim3(H / 128, M / 128), 256, 0, stream>>>(Ob, Wob, bo, out,
                                                               (int)M, (int)H, (int)H);
}
